// AKTModel_71365176591005
// MI455X (gfx1250) — matrix operand order repaired and checked by reading, not compiled, not run on hardware
//
#include <hip/hip_runtime.h>
#include <hip/hip_bf16.h>

#define B_   8
#define LQ_  512
#define LS_  511
#define LP_  512
#define D_   256
#define H_   8
#define DK_  32
#define FF_  1024
#define NL_  4
#define NEGB (-1.0e9f)
#define SCL_ 0.17677669529663689f   /* 1/sqrt(32) */

typedef __attribute__((ext_vector_type(16))) _Float16 v16h;
typedef __attribute__((ext_vector_type(8)))  _Float16 v8h;
typedef __attribute__((ext_vector_type(8)))  float    v8f;

// ---------------- device helpers ----------------

__device__ inline float wave_sum(float v){
#pragma unroll
  for (int m = 16; m >= 1; m >>= 1) v += __shfl_xor(v, m, 32);
  return v;
}
// reduce within each 16-lane half (lanes 0-15 and 16-31 independently)
__device__ inline float half_sum(float v){
#pragma unroll
  for (int m = 8; m >= 1; m >>= 1) v += __shfl_xor(v, m, 32);
  return v;
}
__device__ inline float half_max(float v){
#pragma unroll
  for (int m = 8; m >= 1; m >>= 1) v = fmaxf(v, __shfl_xor(v, m, 32));
  return v;
}
__device__ inline float gelu_exact(float x){
  return 0.5f * x * (1.0f + erff(x * 0.7071067811865475f));
}

// A fragment (16x32, f16): lane holds row m=lane%16.
// lanes 0-15: K = 0..15 (elts 0..15); lanes 16-31: K = 16..31.
// `row` must already point at (m, k0 + kbA) where kbA = (lane>=16)*16.
__device__ inline v16h fragA_f32(const float* row){
  v16h r;
#pragma unroll
  for (int j = 0; j < 8; ++j){ r[j] = (_Float16)row[j]; r[8+j] = (_Float16)row[8+j]; }
  return r;
}
__device__ inline v16h fragA_f16(const _Float16* row){
  v8h lo = *(const v8h*)row;
  v8h hi = *(const v8h*)(row + 8);
  v16h r;
#pragma unroll
  for (int j = 0; j < 8; ++j){ r[j] = lo[j]; r[8+j] = hi[j]; }
  return r;
}
// B fragment (32x16, f16): lane holds column n=lane%16, 16 contiguous K values
// (lanes 0-15: K0..15, lanes 16-31: K16..31). `p` points at (n, k0 + kbB),
// kbB = (lane>=16)*16. Single aligned 32-byte load.
__device__ inline v16h fragB_f16(const _Float16* p){ return *(const v16h*)p; }
__device__ inline v16h fragB_f32(const float* p){
  v16h r;
#pragma unroll
  for (int j = 0; j < 16; ++j) r[j] = (_Float16)p[j];
  return r;
}

#define WMMA(a,b,c) __builtin_amdgcn_wmma_f32_16x16x32_f16(false,(a),false,(b),(short)0,(c),false,false)

// ---------------- kernels ----------------

// weight transpose-convert: src fp32 (K x N) row-major -> dst f16 (N x K)
__global__ void k_wconv(const float* __restrict__ src, _Float16* __restrict__ dst,
                        int K, int N){
  int idx = blockIdx.x * blockDim.x + threadIdx.x;
  if (idx >= K * N) return;
  int n = idx % N, k = idx / N;
  dst[(size_t)n * K + k] = (_Float16)src[idx];
}

// C(MxN) = act(A(MxK fp32) @ W + bias), W given transposed f16 (N x K).
// one wave computes a 16x64 tile.
__global__ void k_gemm(const float* __restrict__ A, int lda,
                       const _Float16* __restrict__ Wt,
                       const float* __restrict__ bias,
                       float* __restrict__ C,
                       int M, int K, int N, int act){
  int gw   = blockIdx.x * (blockDim.x >> 5) + (threadIdx.x >> 5);
  int lane = threadIdx.x & 31;
  int tilesN = N >> 6;
  if (gw >= (M >> 4) * tilesN) return;
  int mt = gw / tilesN, nt = gw % tilesN;
  int m0 = mt << 4, n0 = nt << 6;
  int half = lane >> 4, r = lane & 15;
  const float*    arow = A  + (size_t)(m0 + r) * lda + half * 16;
  const _Float16* b0p  = Wt + (size_t)(n0 + r) * K + half * 16;
  const _Float16* b1p  = b0p + (size_t)16 * K;
  const _Float16* b2p  = b0p + (size_t)32 * K;
  const _Float16* b3p  = b0p + (size_t)48 * K;
  v8f acc[4] = {};
  for (int k0 = 0; k0 < K; k0 += 32){
    v16h a = fragA_f32(arow + k0);
    acc[0] = WMMA(a, fragB_f16(b0p + k0), acc[0]);
    acc[1] = WMMA(a, fragB_f16(b1p + k0), acc[1]);
    acc[2] = WMMA(a, fragB_f16(b2p + k0), acc[2]);
    acc[3] = WMMA(a, fragB_f16(b3p + k0), acc[3]);
  }
#pragma unroll
  for (int t = 0; t < 4; ++t){
    int n = n0 + t * 16 + r;
    float bs = bias ? bias[n] : 0.0f;
#pragma unroll
    for (int i = 0; i < 8; ++i){
      float v = acc[t][i] + bs;
      if (act == 1) v = gelu_exact(v);
      C[(size_t)(m0 + i + half * 8) * N + n] = v;
    }
  }
}

// Fused scores + softmax statistics (no materialized fp32 score tensor).
// One wave per (b,h,q-tile). DK=32 -> one WMMA per 16x16 score tile.
// Pass 1: row maxima (registers + half-wave shfl reduce).
// Pass 2: recompute tiles, emit unnormalized exp(v-max) as f16 into A16,
//         accumulate row sums -> rowsum. Normalization happens in k_av.
__global__ void k_attn_scores(const float* __restrict__ Q, const float* __restrict__ Kb,
                              const float* __restrict__ lg,
                              _Float16* __restrict__ A16, float* __restrict__ rowsum,
                              int Lvalid){
  const int QT = LP_ / 16;
  int gw = blockIdx.x * (blockDim.x >> 5) + (threadIdx.x >> 5);
  if (gw >= B_ * H_ * QT) return;
  int lane = threadIdx.x & 31;
  int qt = gw % QT, bh = gw / QT;
  int h = bh % H_, b = bh / H_;
  float gamma = log1pf(expf(lg[h]));
  int half = lane >> 4, r = lane & 15;
  int q0 = qt * 16;
  v16h a = fragA_f32(Q + (size_t)(b * LP_ + q0 + r) * D_ + h * DK_ + half * 16);
  const float* kbase = Kb + (size_t)b * LP_ * D_ + h * DK_ + half * 16;

  float mx[8];
#pragma unroll
  for (int i = 0; i < 8; ++i) mx[i] = NEGB;
  // pass 1: maxima
  for (int st = 0; st < QT; ++st){
    v16h bb = fragB_f32(kbase + (size_t)(st * 16 + r) * D_);
    v8f c = {};
    c = WMMA(a, bb, c);
    int s = st * 16 + r;
#pragma unroll
    for (int i = 0; i < 8; ++i){
      int q = q0 + i + half * 8;
      float bias = (q >= s ? 0.0f : NEGB) - gamma * fmaxf((float)(q - s), 0.0f);
      if (s >= Lvalid) bias = NEGB;
      mx[i] = fmaxf(mx[i], c[i] * SCL_ + bias);
    }
  }
#pragma unroll
  for (int i = 0; i < 8; ++i) mx[i] = half_max(mx[i]);

  // pass 2: exp + sums, write f16 A
  float sm[8];
#pragma unroll
  for (int i = 0; i < 8; ++i) sm[i] = 0.0f;
  _Float16* arow16 = A16 + (size_t)bh * LP_ * LP_;
  for (int st = 0; st < QT; ++st){
    v16h bb = fragB_f32(kbase + (size_t)(st * 16 + r) * D_);
    v8f c = {};
    c = WMMA(a, bb, c);
    int s = st * 16 + r;
#pragma unroll
    for (int i = 0; i < 8; ++i){
      int q = q0 + i + half * 8;
      float bias = (q >= s ? 0.0f : NEGB) - gamma * fmaxf((float)(q - s), 0.0f);
      if (s >= Lvalid) bias = NEGB;
      float e = expf(c[i] * SCL_ + bias - mx[i]);
      sm[i] += e;
      arow16[(size_t)q * LP_ + s] = (_Float16)e;
    }
  }
#pragma unroll
  for (int i = 0; i < 8; ++i) sm[i] = half_sum(sm[i]);
  if (r == 0){
#pragma unroll
    for (int i = 0; i < 8; ++i) rowsum[bh * LP_ + q0 + i + half * 8] = sm[i];
  }
}

// V (b,t,h*dk fp32) -> Vt f16 [(b*H+h)*DK+dk][t]
__global__ void k_vtrans(const float* __restrict__ V, _Float16* __restrict__ Vt){
  int idx = blockIdx.x * blockDim.x + threadIdx.x;
  if (idx >= B_ * LP_ * D_) return;
  int d = idx % D_, t = (idx / D_) % LP_, b = idx / (D_ * LP_);
  int h = d / DK_, dk = d % DK_;
  Vt[((size_t)((b * H_ + h) * DK_ + dk)) * LP_ + t] = (_Float16)V[idx];
}

// O[b,q,h,:] = (exp-A[b,h,q,:] @ V[b,:,h,:]) / rowsum[q]
// one wave per (b,h,q-tile): 16x32 out.
__global__ void k_av(const _Float16* __restrict__ A16, const _Float16* __restrict__ Vt,
                     const float* __restrict__ rowsum, float* __restrict__ O){
  const int QT = LP_ / 16;
  int gw = blockIdx.x * (blockDim.x >> 5) + (threadIdx.x >> 5);
  if (gw >= B_ * H_ * QT) return;
  int lane = threadIdx.x & 31;
  int qt = gw % QT, bh = gw / QT;
  int h = bh % H_, b = bh / H_;
  int half = lane >> 4, r = lane & 15;
  int q0 = qt * 16;
  const _Float16* arow = A16 + (size_t)(bh * LP_ + q0 + r) * LP_ + half * 16;
  const _Float16* v0p  = Vt + (size_t)(bh * DK_ + r) * LP_ + half * 16;
  const _Float16* v1p  = v0p + (size_t)16 * LP_;
  v8f acc0 = {}, acc1 = {};
  for (int k0 = 0; k0 < LP_; k0 += 32){
    v16h a = fragA_f16(arow + k0);
    acc0 = WMMA(a, fragB_f16(v0p + k0), acc0);
    acc1 = WMMA(a, fragB_f16(v1p + k0), acc1);
  }
#pragma unroll
  for (int i = 0; i < 8; ++i){
    int q = q0 + i + half * 8;
    float inv = 1.0f / rowsum[bh * LP_ + q];
    float* orow = O + (size_t)(b * LP_ + q) * D_ + h * DK_;
    orow[r]      = acc0[i] * inv;
    orow[16 + r] = acc1[i] * inv;
  }
}

// x = LN(x + add) * g + b ; one wave per 256-wide row
__global__ void k_addln(float* __restrict__ x, const float* __restrict__ add,
                        const float* __restrict__ g, const float* __restrict__ bt){
  int row = blockIdx.x * (blockDim.x >> 5) + (threadIdx.x >> 5);
  if (row >= B_ * LP_) return;
  int lane = threadIdx.x & 31;
  float v[8], s = 0.0f;
#pragma unroll
  for (int j = 0; j < 8; ++j){
    int c = j * 32 + lane;
    v[j] = x[(size_t)row * D_ + c] + add[(size_t)row * D_ + c];
    s += v[j];
  }
  float mu = wave_sum(s) * (1.0f / D_);
  float q = 0.0f;
#pragma unroll
  for (int j = 0; j < 8; ++j){ float d = v[j] - mu; q += d * d; }
  float inv = rsqrtf(wave_sum(q) * (1.0f / D_) + 1e-6f);
#pragma unroll
  for (int j = 0; j < 8; ++j){
    int c = j * 32 + lane;
    x[(size_t)row * D_ + c] = (v[j] - mu) * inv * g[c] + bt[c];
  }
}

__global__ void k_build_q(const int* __restrict__ qs, const float* __restrict__ qemb,
                          const float* __restrict__ posq, float* __restrict__ xq){
  int idx = blockIdx.x * blockDim.x + threadIdx.x;
  if (idx >= B_ * LP_ * D_) return;
  int d = idx % D_, t = (idx / D_) % LP_, b = idx / (D_ * LP_);
  xq[idx] = qemb[(size_t)qs[b * LQ_ + t] * D_ + d] + posq[t * D_ + d];
}

// concat [q_emb(256) | diff*r(64) | resp(64) | feat@fw+fb(128)] -> cat (b,t,512)
__global__ void k_build_cat(const int* __restrict__ iq, const int* __restrict__ ir,
                            const float* __restrict__ feat,
                            const float* __restrict__ qemb, const float* __restrict__ demb,
                            const float* __restrict__ remb,
                            const float* __restrict__ fw, const float* __restrict__ fb,
                            float* __restrict__ cat){
  int idx = blockIdx.x * blockDim.x + threadIdx.x;
  if (idx >= B_ * LP_ * 512) return;
  int c = idx % 512, t = (idx / 512) % LP_, b = idx / (512 * LP_);
  float v = 0.0f;
  if (t < LS_){
    int q = iq[b * LS_ + t], rr = ir[b * LS_ + t];
    if (c < 256)      v = qemb[(size_t)q * 256 + c];
    else if (c < 320) v = demb[(size_t)q * 64 + (c - 256)] * (float)rr;
    else if (c < 384) v = remb[rr * 64 + (c - 320)];
    else {
      int j = c - 384; float a = fb[j];
      const float* f = feat + (size_t)(b * LS_ + t) * 8;
#pragma unroll
      for (int k = 0; k < 8; ++k) a += f[k] * fw[k * 128 + j];
      v = a;
    }
  }
  cat[idx] = v;
}

__global__ void k_post_s(float* __restrict__ xs, const float* __restrict__ poss,
                         const float* __restrict__ feat,
                         const float* __restrict__ tw, const float* __restrict__ tb){
  int idx = blockIdx.x * blockDim.x + threadIdx.x;
  if (idx >= B_ * LP_ * D_) return;
  int d = idx % D_, t = (idx / D_) % LP_, b = idx / (D_ * LP_);
  if (t < LS_){
    float te = tanhf(feat[(size_t)(b * LS_ + t) * 8] * tw[d] + tb[d]);
    xs[idx] += poss[t * D_ + d] + te;
  } else xs[idx] = 0.0f;
}

__global__ void k_shift(const float* __restrict__ xq, float* __restrict__ xc,
                        float* __restrict__ xk){
  int idx = blockIdx.x * blockDim.x + threadIdx.x;
  if (idx >= B_ * LP_ * D_) return;
  int d = idx % D_, t = (idx / D_) % LP_, b = idx / (D_ * LP_);
  float c = 0.0f, k = 0.0f;
  if (t < LS_){
    c = xq[(size_t)(b * LP_ + t + 1) * D_ + d];
    k = xq[(size_t)(b * LP_ + t) * D_ + d];
  }
  xc[idx] = c; xk[idx] = k;
}

__global__ void k_headdot(const float* __restrict__ Hd, const float* __restrict__ w2,
                          const float* __restrict__ b2, float* __restrict__ out){
  int row = blockIdx.x * (blockDim.x >> 5) + (threadIdx.x >> 5);
  if (row >= B_ * LS_) return;
  int lane = threadIdx.x & 31;
  int b = row / LS_, t = row % LS_;
  const float* hr = Hd + (size_t)(b * LP_ + t) * D_;
  float s = 0.0f;
#pragma unroll
  for (int j = 0; j < 8; ++j){ int c = j * 32 + lane; s += hr[c] * w2[c]; }
  s = wave_sum(s);
  if (lane == 0) out[row] = s + b2[0];
}

// ---------------- host ----------------

struct StackP {
  const float *Wk,*Wo,*Wq,*Wv,*bk,*bo,*bq,*bv,*fb1,*fb2,*fw1,*fw2,*ln1b,*ln1g,*ln2b,*ln2g,*lg;
};
static StackP mk_stack(void* const* d_in, int base){
  StackP s;
  s.Wk =(const float*)d_in[base+0];  s.Wo =(const float*)d_in[base+1];
  s.Wq =(const float*)d_in[base+2];  s.Wv =(const float*)d_in[base+3];
  s.bk =(const float*)d_in[base+4];  s.bo =(const float*)d_in[base+5];
  s.bq =(const float*)d_in[base+6];  s.bv =(const float*)d_in[base+7];
  s.fb1=(const float*)d_in[base+8];  s.fb2=(const float*)d_in[base+9];
  s.fw1=(const float*)d_in[base+10]; s.fw2=(const float*)d_in[base+11];
  s.ln1b=(const float*)d_in[base+12]; s.ln1g=(const float*)d_in[base+13];
  s.ln2b=(const float*)d_in[base+14]; s.ln2g=(const float*)d_in[base+15];
  s.lg =(const float*)d_in[base+16];
  return s;
}
struct StackW { _Float16 *Wq[NL_],*Wk[NL_],*Wv[NL_],*Wo[NL_],*W1[NL_],*W2[NL_]; };

extern "C" void kernel_launch(void* const* d_in, const int* in_sizes, int n_in,
                              void* d_out, int out_size, void* d_ws, size_t ws_size,
                              hipStream_t stream){
  (void)in_sizes; (void)n_in; (void)out_size; (void)ws_size;
  // pytree (sorted-key) flatten order of setup_inputs
  const int*   questions = (const int*)  d_in[0];
  const int*   iq        = (const int*)  d_in[1];
  const int*   ir        = (const int*)  d_in[2];
  const float* feat      = (const float*)d_in[3];
  const float* diff_emb  = (const float*)d_in[4];
  const float* fpb       = (const float*)d_in[5];
  const float* fpw       = (const float*)d_in[6];
  const float* ipb       = (const float*)d_in[7];
  const float* ipw       = (const float*)d_in[8];
  StackP stKR = mk_stack(d_in, 9);
  const float* out_b1    = (const float*)d_in[26];
  const float* out_b2    = (const float*)d_in[27];
  const float* out_w1    = (const float*)d_in[28];
  const float* out_w2    = (const float*)d_in[29];
  const float* pos_q     = (const float*)d_in[30];
  const float* pos_s     = (const float*)d_in[31];
  StackP stQ  = mk_stack(d_in, 32);
  const float* qemb      = (const float*)d_in[49];
  const float* remb      = (const float*)d_in[50];
  StackP stS  = mk_stack(d_in, 51);
  const float* tpb       = (const float*)d_in[68];
  const float* tpw       = (const float*)d_in[69];

  const int MTi = B_ * LP_;                 // 4096 padded rows
  float* ws = (float*)d_ws;
  const size_t XN = (size_t)MTi * D_;       // 1,048,576 floats
  float* xq = ws; ws += XN;
  float* xs = ws; ws += XN;
  float* xc = ws; ws += XN;
  float* xk = ws; ws += XN;
  float* Qb = ws; ws += XN;
  float* Kb = ws; ws += XN;
  float* Vb = ws; ws += XN;
  float* t1 = ws; ws += (size_t)MTi * FF_;  // FFN hidden / concat buffer
  float* rowsum = ws; ws += (size_t)B_ * H_ * LP_;
  _Float16* A16 = (_Float16*)ws; ws += (size_t)B_ * H_ * LP_ * LP_ / 2;
  _Float16* Vt  = (_Float16*)ws; ws += XN / 2;
  _Float16* wptr = (_Float16*)ws;

  auto conv = [&](const float* src, int K, int N) -> _Float16* {
    _Float16* dst = wptr; wptr += (size_t)K * N;
    int n = K * N;
    k_wconv<<<(n + 255) / 256, 256, 0, stream>>>(src, dst, K, N);
    return dst;
  };
  auto convStack = [&](const StackP& st) -> StackW {
    StackW w;
    for (int l = 0; l < NL_; ++l){
      w.Wq[l] = conv(st.Wq  + (size_t)l * D_ * D_,  D_,  D_);
      w.Wk[l] = conv(st.Wk  + (size_t)l * D_ * D_,  D_,  D_);
      w.Wv[l] = conv(st.Wv  + (size_t)l * D_ * D_,  D_,  D_);
      w.Wo[l] = conv(st.Wo  + (size_t)l * D_ * D_,  D_,  D_);
      w.W1[l] = conv(st.fw1 + (size_t)l * D_ * FF_, D_,  FF_);
      w.W2[l] = conv(st.fw2 + (size_t)l * FF_ * D_, FF_, D_);
    }
    return w;
  };
  auto gemm = [&](const float* A, int lda, const _Float16* Wt, const float* bias,
                  float* C, int M, int K, int N, int act){
    int waves = (M / 16) * (N / 64);
    k_gemm<<<(waves + 3) / 4, 128, 0, stream>>>(A, lda, Wt, bias, C, M, K, N, act);
  };

  // convert all weights to transposed f16 once
  _Float16* ipw16 = conv(ipw, 512, D_);
  _Float16* ow1   = conv(out_w1, D_, D_);
  StackW swQ  = convStack(stQ);
  StackW swS  = convStack(stS);
  StackW swKR = convStack(stKR);

  const int EW = B_ * LP_ * D_;             // elementwise count
  const int AW = B_ * H_ * (LP_ / 16);      // attention waves
  const int LR = B_ * LP_;                  // LN rows

  auto attn_layer = [&](float* x, const float* kin, const float* vin,
                        const StackP& st, const StackW& sw, int l, int Lvalid){
    gemm(x,   D_, sw.Wq[l], st.bq + l * D_, Qb, MTi, D_, D_, 0);
    gemm(kin, D_, sw.Wk[l], st.bk + l * D_, Kb, MTi, D_, D_, 0);
    gemm(vin, D_, sw.Wv[l], st.bv + l * D_, Vb, MTi, D_, D_, 0);
    k_vtrans<<<(EW + 255) / 256, 256, 0, stream>>>(Vb, Vt);
    k_attn_scores<<<(AW + 3) / 4, 128, 0, stream>>>(Qb, Kb, st.lg + l * H_,
                                                    A16, rowsum, Lvalid);
    k_av<<<(AW + 3) / 4, 128, 0, stream>>>(A16, Vt, rowsum, Qb);
    gemm(Qb, D_, sw.Wo[l], st.bo + l * D_, Kb, MTi, D_, D_, 0);
    k_addln<<<(LR + 3) / 4, 128, 0, stream>>>(x, Kb, st.ln1g + l * D_, st.ln1b + l * D_);
    gemm(x,  D_,  sw.W1[l], st.fb1 + l * FF_, t1, MTi, D_,  FF_, 1);
    gemm(t1, FF_, sw.W2[l], st.fb2 + l * D_,  Vb, MTi, FF_, D_,  0);
    k_addln<<<(LR + 3) / 4, 128, 0, stream>>>(x, Vb, st.ln2g + l * D_, st.ln2b + l * D_);
  };

  // --- build input representations ---
  k_build_q  <<<(EW + 255) / 256, 256, 0, stream>>>(questions, qemb, pos_q, xq);
  k_build_cat<<<(B_ * LP_ * 512 + 255) / 256, 256, 0, stream>>>(
      iq, ir, feat, qemb, diff_emb, remb, fpw, fpb, t1);
  gemm(t1, 512, ipw16, ipb, xs, MTi, 512, D_, 0);
  k_post_s   <<<(EW + 255) / 256, 256, 0, stream>>>(xs, pos_s, feat, tpw, tpb);

  // --- self-attention encoders ---
  for (int l = 0; l < NL_; ++l) attn_layer(xq, xq, xq, stQ, swQ, l, LP_);
  for (int l = 0; l < NL_; ++l) attn_layer(xs, xs, xs, stS, swS, l, LS_);

  // --- knowledge-retrieval cross-attention ---
  k_shift<<<(EW + 255) / 256, 256, 0, stream>>>(xq, xc, xk);
  for (int l = 0; l < NL_; ++l) attn_layer(xc, xk, xs, stKR, swKR, l, LS_);

  // --- output head ---
  gemm(xc, D_, ow1, out_b1, Qb, MTi, D_, D_, 1);
  int hrows = B_ * LS_;
  k_headdot<<<(hrows + 3) / 4, 128, 0, stream>>>(Qb, out_w2, out_b2, (float*)d_out);
}
